// StructureAE_38199439131012
// MI455X (gfx1250) — matrix operand order repaired and checked by reading, not compiled, not run on hardware
//
#include <hip/hip_runtime.h>
#include <hip/hip_bf16.h>
#include <cstdint>

typedef __attribute__((ext_vector_type(16))) _Float16 v16h;
typedef __attribute__((ext_vector_type(8)))  _Float16 v8h;
typedef __attribute__((ext_vector_type(8)))  float    v8f;

#define N_NODES   16384
#define N_EDGES   524288
#define IN_DIM    256
#define EMBED_DIM 128
#define OUT_DIM   64
#define NEG_SLOPE 0.2f

// ---------------------------------------------------------------------------
// WMMA fragment loaders for V_WMMA_F32_16X16X32_F16 (wave32, ISA 7.12.2)
// A: 16(M) x 32(K) f16. Lane L (l=L&15, hi=L>>4) holds row M=l:
//    halfs j=0..7  -> K = kbase + 8*hi + j        (one 16B load)
//    halfs j=8..15 -> K = kbase + 8*hi + 16 + j-8 (one 16B load)
// B: 32(K) x 16(N) f16 = W^T with W row-major [N x K]. Lane L holds col N=l:
//    halfs j=0..7  -> K = kbase + 8*hi + j; j=8..15 -> K = kbase + 8*hi + 16 + j-8 (two 16B loads)
// ---------------------------------------------------------------------------
__device__ __forceinline__ v16h load_a_frag(const _Float16* __restrict__ A, int lda,
                                            int row0, int kbase, int lane) {
  int l = lane & 15, hi = lane >> 4;
  const _Float16* p = A + (size_t)(row0 + l) * lda + kbase + hi * 8;
  v8h lo = *(const v8h*)(p);
  v8h hh = *(const v8h*)(p + 16);
  v16h f;
#pragma unroll
  for (int j = 0; j < 8; ++j) { f[j] = lo[j]; f[j + 8] = hh[j]; }
  return f;
}

__device__ __forceinline__ v16h load_b_frag(const _Float16* __restrict__ W, int ldw,
                                            int col0, int kbase, int lane) {
  int l = lane & 15, hi = lane >> 4;
  const _Float16* p = W + (size_t)(col0 + l) * ldw + kbase + hi * 8;
  v8h lo = *(const v8h*)(p);
  v8h hh = *(const v8h*)(p + 16);
  v16h f;
#pragma unroll
  for (int j = 0; j < 8; ++j) { f[j] = lo[j]; f[j + 8] = hh[j]; }
  return f;
}

// ---------------------------------------------------------------------------
// Generic row-strip WMMA GEMM: O[M x NDIM] = act(A[M x K] @ W[NDIM x K]^T + bias)
// 256 threads = 8 waves; each wave owns one 16-row strip, all NTILES col tiles.
// ---------------------------------------------------------------------------
template <int KDIM, int NTILES, bool RELU, bool HALF_OUT>
__global__ __launch_bounds__(256) void k_gemm(const _Float16* __restrict__ A,
                                              const _Float16* __restrict__ W,
                                              const float* __restrict__ bias,
                                              void* __restrict__ Ovoid) {
  const int NDIM = NTILES * 16;
  int wave = threadIdx.x >> 5, lane = threadIdx.x & 31;
  int l = lane & 15, hi = lane >> 4;
  int row0 = (blockIdx.x * 8 + wave) * 16;

  v8f acc[NTILES];
#pragma unroll
  for (int nt = 0; nt < NTILES; ++nt) {
    float bv = bias ? bias[nt * 16 + l] : 0.0f;
#pragma unroll
    for (int r = 0; r < 8; ++r) acc[nt][r] = bv;
  }

#pragma unroll
  for (int kb = 0; kb < KDIM; kb += 32) {
    v16h a = load_a_frag(A, KDIM, row0, kb, lane);
#pragma unroll
    for (int nt = 0; nt < NTILES; ++nt) {
      v16h b = load_b_frag(W, KDIM, nt * 16, kb, lane);
      acc[nt] = __builtin_amdgcn_wmma_f32_16x16x32_f16(false, a, false, b,
                                                       (short)0, acc[nt], false, false);
    }
  }

#pragma unroll
  for (int nt = 0; nt < NTILES; ++nt) {
#pragma unroll
    for (int r = 0; r < 8; ++r) {
      float v = acc[nt][r];
      if (RELU) v = v > 0.0f ? v : 0.0f;
      size_t off = (size_t)(row0 + r + hi * 8) * NDIM + nt * 16 + l;
      if (HALF_OUT) ((_Float16*)Ovoid)[off] = (_Float16)v;
      else          ((float*)Ovoid)[off]    = v;
    }
  }
}

// ---------------------------------------------------------------------------
// recon = sigmoid(E @ E^T), E f16 [16384 x 64].  Store-BW bound: 1.07 GB out.
// Workgroup = 128x128 tile (8 waves x one 16x128 strip); K=64 = 2 WMMA steps.
// ---------------------------------------------------------------------------
__global__ __launch_bounds__(256) void k_recon(const _Float16* __restrict__ Eh,
                                               float* __restrict__ out) {
  int wave = threadIdx.x >> 5, lane = threadIdx.x & 31;
  int l = lane & 15, hi = lane >> 4;
  int row0 = (blockIdx.y * 8 + wave) * 16;
  int col0 = blockIdx.x * 128;

  v16h a0 = load_a_frag(Eh, OUT_DIM, row0, 0, lane);
  v16h a1 = load_a_frag(Eh, OUT_DIM, row0, 32, lane);

#pragma unroll
  for (int nt = 0; nt < 8; ++nt) {
    v16h b0 = load_b_frag(Eh, OUT_DIM, col0 + nt * 16, 0, lane);
    v16h b1 = load_b_frag(Eh, OUT_DIM, col0 + nt * 16, 32, lane);
    v8f acc;
#pragma unroll
    for (int r = 0; r < 8; ++r) acc[r] = 0.0f;
    acc = __builtin_amdgcn_wmma_f32_16x16x32_f16(false, a0, false, b0, (short)0, acc, false, false);
    acc = __builtin_amdgcn_wmma_f32_16x16x32_f16(false, a1, false, b1, (short)0, acc, false, false);
#pragma unroll
    for (int r = 0; r < 8; ++r) {
      float v = 1.0f / (1.0f + __expf(-acc[r]));
      out[(size_t)(row0 + r + hi * 8) * N_NODES + col0 + nt * 16 + l] = v;
    }
  }
}

// ---------------------------------------------------------------------------
// Small helper kernels
// ---------------------------------------------------------------------------
__global__ void k_f32_to_f16(const float* __restrict__ in, _Float16* __restrict__ out, int n) {
  int i = blockIdx.x * blockDim.x + threadIdx.x;
  if (i < n) out[i] = (_Float16)in[i];
}

__global__ void k_alpha(const float* __restrict__ g, const float* __restrict__ a_src,
                        const float* __restrict__ a_dst, float* __restrict__ as,
                        float* __restrict__ ad, int n) {
  int i = blockIdx.x * blockDim.x + threadIdx.x;
  if (i >= n) return;
  const float* gr = g + (size_t)i * OUT_DIM;
  float s = 0.f, d = 0.f;
#pragma unroll 8
  for (int c = 0; c < OUT_DIM; ++c) { float v = gr[c]; s += v * a_src[c]; d += v * a_dst[c]; }
  as[i] = s; ad[i] = d;
}

__global__ void k_init(float* __restrict__ emax, float* __restrict__ denom,
                       float* __restrict__ embed, const float* __restrict__ b_gat, int n) {
  int i = blockIdx.x * blockDim.x + threadIdx.x;
  if (i < n) { emax[i] = -__builtin_inff(); denom[i] = 0.0f; }
  if (i < n * OUT_DIM) embed[i] = b_gat[i & (OUT_DIM - 1)];
}

__device__ __forceinline__ float leaky(float e) { return e > 0.0f ? e : NEG_SLOPE * e; }

// float atomic max via signed-max / unsigned-min trick (monotone bit patterns)
__device__ __forceinline__ void atomicMaxF(float* addr, float v) {
  if (v >= 0.0f) atomicMax((int*)addr, __float_as_int(v));
  else           atomicMin((unsigned int*)addr, __float_as_uint(v));
}

__device__ __forceinline__ void edge_sd(const long long* __restrict__ ei, int i,
                                        int& s, int& d) {
  if (i < N_EDGES) { s = (int)ei[i]; d = (int)ei[N_EDGES + i]; }
  else             { s = i - N_EDGES; d = s; }  // appended self-loops
}

__global__ void k_edge_max(const long long* __restrict__ ei, const float* __restrict__ as,
                           const float* __restrict__ ad, float* __restrict__ emax) {
  int i = blockIdx.x * blockDim.x + threadIdx.x;
  if (i >= N_EDGES + N_NODES) return;
  int s, d; edge_sd(ei, i, s, d);
  atomicMaxF(&emax[d], leaky(as[s] + ad[d]));
}

__global__ void k_edge_sum(const long long* __restrict__ ei, const float* __restrict__ as,
                           const float* __restrict__ ad, const float* __restrict__ emax,
                           float* __restrict__ denom) {
  int i = blockIdx.x * blockDim.x + threadIdx.x;
  if (i >= N_EDGES + N_NODES) return;
  int s, d; edge_sd(ei, i, s, d);
  atomicAdd(&denom[d], __expf(leaky(as[s] + ad[d]) - emax[d]));
}

// one thread per (edge, 4-column group): 4 float atomic adds, vector g read
__global__ void k_edge_aggr(const long long* __restrict__ ei, const float* __restrict__ as,
                            const float* __restrict__ ad, const float* __restrict__ emax,
                            const float* __restrict__ denom, const float* __restrict__ g,
                            float* __restrict__ embed) {
  int t = blockIdx.x * blockDim.x + threadIdx.x;
  if (t >= (N_EDGES + N_NODES) * (OUT_DIM / 4)) return;
  int i = t / (OUT_DIM / 4);
  int c = (t % (OUT_DIM / 4)) * 4;
  int s, d; edge_sd(ei, i, s, d);
  float w = __expf(leaky(as[s] + ad[d]) - emax[d]) / denom[d];
  float4 gv = *(const float4*)(g + (size_t)s * OUT_DIM + c);
  float* ep = embed + (size_t)d * OUT_DIM + c;
  atomicAdd(ep + 0, w * gv.x);
  atomicAdd(ep + 1, w * gv.y);
  atomicAdd(ep + 2, w * gv.z);
  atomicAdd(ep + 3, w * gv.w);
}

// ---------------------------------------------------------------------------
extern "C" void kernel_launch(void* const* d_in, const int* in_sizes, int n_in,
                              void* d_out, int out_size, void* d_ws, size_t ws_size,
                              hipStream_t stream) {
  const float*     x       = (const float*)d_in[0];
  const long long* ei      = (const long long*)d_in[1];   // int64 edge_index [2, E]
  const float*     W_dense = (const float*)d_in[2];
  const float*     b_dense = (const float*)d_in[3];
  const float*     W_gat   = (const float*)d_in[4];
  const float*     a_src   = (const float*)d_in[5];
  const float*     a_dst   = (const float*)d_in[6];
  const float*     b_gat   = (const float*)d_in[7];

  float* recon = (float*)d_out;                               // [N, N]
  float* embed = recon + (size_t)N_NODES * N_NODES;           // [N, OUT_DIM] fp32 output

  // carve workspace (256B aligned slices)
  char* w = (char*)d_ws;
  auto carve = [&](size_t bytes) { char* p = w; w += (bytes + 255) & ~(size_t)255; return p; };
  _Float16* x16  = (_Float16*)carve((size_t)N_NODES * IN_DIM * 2);
  _Float16* Wd16 = (_Float16*)carve((size_t)EMBED_DIM * IN_DIM * 2);
  _Float16* Wg16 = (_Float16*)carve((size_t)OUT_DIM * EMBED_DIM * 2);
  _Float16* h16  = (_Float16*)carve((size_t)N_NODES * EMBED_DIM * 2);
  float*    gbuf = (float*)   carve((size_t)N_NODES * OUT_DIM * 4);
  float*    as   = (float*)   carve((size_t)N_NODES * 4);
  float*    ad   = (float*)   carve((size_t)N_NODES * 4);
  float*    emax = (float*)   carve((size_t)N_NODES * 4);
  float*    den  = (float*)   carve((size_t)N_NODES * 4);
  _Float16* e16  = (_Float16*)carve((size_t)N_NODES * OUT_DIM * 2);

  const int B = 256;
  auto cdiv = [](long long a, long long b) { return (int)((a + b - 1) / b); };

  // fp32 -> f16 operand conversion (weights are tiny; x is 16 MB -> 8 MB)
  k_f32_to_f16<<<cdiv((long long)N_NODES * IN_DIM, B), B, 0, stream>>>(x, x16, N_NODES * IN_DIM);
  k_f32_to_f16<<<cdiv(EMBED_DIM * IN_DIM, B), B, 0, stream>>>(W_dense, Wd16, EMBED_DIM * IN_DIM);
  k_f32_to_f16<<<cdiv(OUT_DIM * EMBED_DIM, B), B, 0, stream>>>(W_gat, Wg16, OUT_DIM * EMBED_DIM);

  // h = relu(x @ W_dense^T + b_dense)  [WMMA, f16 out]
  k_gemm<IN_DIM, EMBED_DIM / 16, true, true>
      <<<N_NODES / 128, 256, 0, stream>>>(x16, Wd16, b_dense, h16);
  // g = h @ W_gat^T  [WMMA, f32 out]
  k_gemm<EMBED_DIM, OUT_DIM / 16, false, false>
      <<<N_NODES / 128, 256, 0, stream>>>(h16, Wg16, nullptr, gbuf);

  // attention logits per node
  k_alpha<<<cdiv(N_NODES, B), B, 0, stream>>>(gbuf, a_src, a_dst, as, ad, N_NODES);

  // init emax/-inf, denom/0, embed/b_gat
  k_init<<<cdiv(N_NODES * OUT_DIM, B), B, 0, stream>>>(emax, den, embed, b_gat, N_NODES);

  // segment softmax + aggregation over E+N edges (self-loops appended)
  int tot = N_EDGES + N_NODES;
  k_edge_max <<<cdiv(tot, B), B, 0, stream>>>(ei, as, ad, emax);
  k_edge_sum <<<cdiv(tot, B), B, 0, stream>>>(ei, as, ad, emax, den);
  k_edge_aggr<<<cdiv((long long)tot * (OUT_DIM / 4), B), B, 0, stream>>>(ei, as, ad, emax, den, gbuf, embed);

  // embed fp32 -> f16 for the big WMMA
  k_f32_to_f16<<<cdiv(N_NODES * OUT_DIM, B), B, 0, stream>>>(embed, e16, N_NODES * OUT_DIM);

  // recon = sigmoid(E @ E^T)  [store-bandwidth bound]
  dim3 rg(N_NODES / 128, N_NODES / 128);
  k_recon<<<rg, 256, 0, stream>>>(e16, recon);
}
